// GraphBlock_52158082842829
// MI455X (gfx1250) — compile-verified, matrix operand order checked by reading, not run on hardware
//
#include <hip/hip_runtime.h>
#include <hip/hip_bf16.h>

typedef __attribute__((ext_vector_type(16))) _Float16 v16h;
typedef __attribute__((ext_vector_type(8)))  float    v8f;

#define DCH 128            // feature dim
#define NHEAD 4

// ---------------------------------------------------------------------------
// Pack the three 128x128 f32 weight matrices into f16, laid out in the exact
// per-lane WMMA B-fragment order:
//   half index = ((((m*8 + ct)*4 + kc)*32 + lane)*16 + i
// so the GEMM kernel can load each B fragment as one contiguous 32B v16h.
// ISA 16-bit 32x16 B layout: element i (vgpr v=i>>1, pos=i&1):
//   K = kc*32 + 16*(v>=4) + 8*(lane>=16) + 2*(v&3) + pos ; N = ct*16 + (lane&15)
// ---------------------------------------------------------------------------
__global__ __launch_bounds__(256) void wpack_kernel(
    const float* __restrict__ w0, const float* __restrict__ w1,
    const float* __restrict__ w2, unsigned int* __restrict__ wpk)
{
    int t = blockIdx.x * 256 + threadIdx.x;     // pair index
    if (t >= 3 * 8 * 4 * 32 * 8) return;        // 24576 pairs
    int j    = t & 7;
    int lane = (t >> 3) & 31;
    int kc   = (t >> 8) & 3;
    int ct   = (t >> 10) & 7;
    int m    = t >> 13;
    int hi   = lane >> 4;
    int k    = kc * 32 + ((j >> 2) << 4) + (hi << 3) + ((j & 3) << 1);
    int ncol = ct * 16 + (lane & 15);
    const float* w = (m == 0) ? w0 : (m == 1) ? w1 : w2;
    union { _Float16 h[2]; unsigned int u; } p;
    p.h[0] = (_Float16)w[k * DCH + ncol];
    p.h[1] = (_Float16)w[(k + 1) * DCH + ncol];
    wpk[t] = p.u;
}

// ---------------------------------------------------------------------------
// Fused 3-matrix GEMM on WMMA: obase = [x@Wl | x@Wr | x@Wgcn] (each n x 128).
// block: 256 threads (8 waves), 128 rows. wave = row-tile (16 rows).
// A fragments in registers straight from x (row index CLAMPED, not masked:
// a garbage A row only affects the same out-of-range D row, never stored),
// reused across 24 tiles (8 col-tiles x 3 mats) = 96 WMMAs/wave.
// B fragments: one contiguous v16h global load each (L2/WGP$-resident).
// Stores: block-uniform `full` fast path -> 8 unguarded b32 stores with
// immediate offsets from one base pointer (no per-lane compares/branches).
// ---------------------------------------------------------------------------
__global__ __launch_bounds__(256) void gemm3_wmma_kernel(
    const float* __restrict__ x, const _Float16* __restrict__ wpk,
    float* __restrict__ obase, int n)
{
    const int wave = threadIdx.x >> 5;
    const int lane = threadIdx.x & 31;
    const int hi   = lane >> 4;
    const long rowbase = (long)blockIdx.x * 128;
    const bool full = (rowbase + 128) <= (long)n;   // block-uniform

    long row = rowbase + wave * 16 + (lane & 15);
    if (row > (long)(n - 1)) row = (long)(n - 1);   // clamp, keep EXEC full
    const float* xrow = x + row * DCH;

    // A fragments: ISA 16-bit 16x32 layout; K pairs contiguous in x rows.
    v16h afrag[4];
#pragma unroll
    for (int kc = 0; kc < 4; ++kc) {
#pragma unroll
        for (int j = 0; j < 8; ++j) {
            int k = kc * 32 + ((j >> 2) << 4) + (hi << 3) + ((j & 3) << 1);
            float2 f = *(const float2*)(xrow + k);
            afrag[kc][2 * j]     = (_Float16)f.x;
            afrag[kc][2 * j + 1] = (_Float16)f.y;
        }
    }

    const _Float16* wlane = wpk + lane * 16;
    const long drow = rowbase + wave * 16 + hi * 8;  // first D row of this lane

#pragma unroll
    for (int m = 0; m < 3; ++m) {
        float* op = obase + (size_t)m * (size_t)n * DCH;
#pragma unroll
        for (int ct = 0; ct < 8; ++ct) {
            v8f acc = {};
#pragma unroll
            for (int kc = 0; kc < 4; ++kc) {
                const v16h bfrag = *(const v16h*)(
                    wlane + (size_t)(((m * 8 + ct) * 4) + kc) * 512);
                acc = __builtin_amdgcn_wmma_f32_16x16x32_f16(
                    false, afrag[kc], false, bfrag, (short)0, acc, false, false);
            }
            // D layout: VGPR j -> M = drow + j, N = ct*16 + (lane&15)
            float* dst = op + drow * DCH + ct * 16 + (lane & 15);
            if (full) {
#pragma unroll
                for (int j = 0; j < 8; ++j) dst[j * DCH] = acc[j];
            } else {
#pragma unroll
                for (int j = 0; j < 8; ++j)
                    if (drow + j < (long)n) dst[j * DCH] = acc[j];
            }
        }
    }
}

// ---------------------------------------------------------------------------
// helpers
// ---------------------------------------------------------------------------
__device__ __forceinline__ void atomicMaxF(float* addr, float val) {
    if (val >= 0.0f) atomicMax((int*)addr, __float_as_int(val));
    else             atomicMin((unsigned int*)addr, __float_as_uint(val));
}

__global__ __launch_bounds__(256) void init_kernel(
    float* acc, float* mmax, float* denom, float* deg, float* sums, long n)
{
    long i = (long)blockIdx.x * 256 + threadIdx.x;
    if (i < n * DCH)   acc[i]   = 0.0f;
    if (i < n * NHEAD) { mmax[i] = -__builtin_inff(); denom[i] = 0.0f; }
    if (i < n)         deg[i]   = 0.0f;
    if (i < 256)       sums[i]  = 0.0f;   // sums[128] + sumsq[128], contiguous
}

// ---------------------------------------------------------------------------
// Pass 1: one wave32 per edge. logits[e,h] = att . leaky_relu(xl[s]+xr[d]);
// atomicMax into per-(dst,head) running max; degree count.
// ---------------------------------------------------------------------------
__global__ __launch_bounds__(256) void edge_logits_kernel(
    const int* __restrict__ ei, int E, long Et,
    const float* __restrict__ xl, const float* __restrict__ xr,
    const float* __restrict__ att,
    float* __restrict__ logits, float* __restrict__ mmax, float* __restrict__ deg)
{
    long e = (long)blockIdx.x * 8 + (threadIdx.x >> 5);
    if (e >= Et) return;                       // uniform per wave
    const int lane = threadIdx.x & 31;
    int s, d;
    if (e < (long)E) { s = ei[e]; d = ei[(long)E + e]; }
    else             { s = d = (int)(e - E); } // self loop

    const float4 xlv = ((const float4*)(xl + (long)s * DCH))[lane];
    const float4 xrv = ((const float4*)(xr + (long)d * DCH))[lane];
    const float4 av  = ((const float4*)att)[lane];
    float p = 0.0f, t;
    t = xlv.x + xrv.x; t = t > 0.0f ? t : 0.2f * t; p += av.x * t;
    t = xlv.y + xrv.y; t = t > 0.0f ? t : 0.2f * t; p += av.y * t;
    t = xlv.z + xrv.z; t = t > 0.0f ? t : 0.2f * t; p += av.z * t;
    t = xlv.w + xrv.w; t = t > 0.0f ? t : 0.2f * t; p += av.w * t;
    // reduce over the 8 lanes of each head (channels c = lane*4.. -> head = lane>>3)
    p += __shfl_xor(p, 1);
    p += __shfl_xor(p, 2);
    p += __shfl_xor(p, 4);
    if ((lane & 7) == 0) {
        int h = lane >> 3;
        logits[e * NHEAD + h] = p;
        atomicMaxF(&mmax[(long)d * NHEAD + h], p);
    }
    if (lane == 0) atomicAdd(&deg[d], 1.0f);
}

__global__ __launch_bounds__(256) void dinv_kernel(const float* deg, float* dinv, int n) {
    int i = blockIdx.x * 256 + threadIdx.x;
    if (i < n) dinv[i] = rsqrtf(fmaxf(deg[i], 1.0f));
}

// Pass 2: one thread per (edge, head): denom[d,h] += exp(logit - max)
__global__ __launch_bounds__(256) void edge_denom_kernel(
    const float* __restrict__ logits, const float* __restrict__ mmax,
    const int* __restrict__ ei, int E, long Et, float* __restrict__ denom)
{
    long t = (long)blockIdx.x * 256 + threadIdx.x;
    if (t >= Et * NHEAD) return;
    long e = t >> 2; int h = (int)(t & 3);
    int d = (e < (long)E) ? ei[(long)E + e] : (int)(e - E);
    float ex = __expf(logits[t] - mmax[(long)d * NHEAD + h]);
    atomicAdd(&denom[(long)d * NHEAD + h], ex);
}

// Pass 3: one wave32 per edge. acc[d] += alpha_h * xl[s] + norm * xw[s]
// (GAT + GCN scatter fused into a single atomic per channel)
__global__ __launch_bounds__(256) void edge_scatter_kernel(
    const int* __restrict__ ei, int E, long Et,
    const float* __restrict__ xl, const float* __restrict__ xw,
    const float* __restrict__ logits, const float* __restrict__ mmax,
    const float* __restrict__ denom, const float* __restrict__ dinv,
    float* __restrict__ acc)
{
    long e = (long)blockIdx.x * 8 + (threadIdx.x >> 5);
    if (e >= Et) return;
    const int lane = threadIdx.x & 31;
    int s, d;
    if (e < (long)E) { s = ei[e]; d = ei[(long)E + e]; }
    else             { s = d = (int)(e - E); }

    int h = lane >> 3;
    float l     = logits[e * NHEAD + h];
    float alpha = __expf(l - mmax[(long)d * NHEAD + h]) / denom[(long)d * NHEAD + h];
    float norm  = dinv[s] * dinv[d];

    float4 xlv = ((const float4*)(xl + (long)s * DCH))[lane];
    float4 xwv = ((const float4*)(xw + (long)s * DCH))[lane];
    float* dst = acc + (long)d * DCH + lane * 4;
    atomicAdd(dst + 0, alpha * xlv.x + norm * xwv.x);
    atomicAdd(dst + 1, alpha * xlv.y + norm * xwv.y);
    atomicAdd(dst + 2, alpha * xlv.z + norm * xwv.z);
    atomicAdd(dst + 3, alpha * xlv.w + norm * xwv.w);
}

// ---------------------------------------------------------------------------
// BN: pre = acc + b_gat + b_gcn + x (residual) stored in place; block-partial
// per-channel sum/sumsq in LDS then one atomic per channel per block.
// ---------------------------------------------------------------------------
__global__ __launch_bounds__(256) void bn_pre_kernel(
    float* __restrict__ acc, const float* __restrict__ x,
    const float* __restrict__ b_gat, const float* __restrict__ b_gcn,
    float* __restrict__ sums, float* __restrict__ sumsq, int n)
{
    __shared__ float ls[256], lq[256];
    const int c    = threadIdx.x & 127;
    const int half = threadIdx.x >> 7;
    const long r0  = (long)blockIdx.x * 64;
    const float bias = b_gat[c] + b_gcn[c];
    float sacc = 0.0f, qacc = 0.0f;
    for (int rr = half; rr < 64; rr += 2) {
        long r = r0 + rr;
        if (r < n) {
            long idx = r * DCH + c;
            float pre = acc[idx] + bias + x[idx];
            acc[idx] = pre;
            sacc += pre;
            qacc += pre * pre;
        }
    }
    ls[threadIdx.x] = sacc; lq[threadIdx.x] = qacc;
    __syncthreads();
    if (half == 0) {
        atomicAdd(&sums[c],  ls[c] + ls[c + 128]);
        atomicAdd(&sumsq[c], lq[c] + lq[c + 128]);
    }
}

__global__ void bn_stats_kernel(const float* sums, const float* sumsq,
                                float* musd, float invn)
{
    int c = threadIdx.x;
    float mu  = sums[c] * invn;
    float var = sumsq[c] * invn - mu * mu;
    musd[c]       = mu;
    musd[128 + c] = rsqrtf(var + 1e-5f);
}

__global__ __launch_bounds__(256) void bn_apply_kernel(
    const float* __restrict__ acc, const float* __restrict__ musd,
    const float* __restrict__ gamma, const float* __restrict__ beta,
    float* __restrict__ out, long total)
{
    long i = (long)blockIdx.x * 256 + threadIdx.x;
    if (i >= total) return;
    int c = (int)(i & 127);
    float v = (acc[i] - musd[c]) * musd[128 + c] * gamma[c] + beta[c];
    out[i] = v > 0.0f ? v : expm1f(v);    // ELU
}

// ---------------------------------------------------------------------------
extern "C" void kernel_launch(void* const* d_in, const int* in_sizes, int n_in,
                              void* d_out, int out_size, void* d_ws, size_t ws_size,
                              hipStream_t stream)
{
    const float* x     = (const float*)d_in[0];
    const float* wgl   = (const float*)d_in[1];
    const float* wgr   = (const float*)d_in[2];
    const float* att   = (const float*)d_in[3];
    const float* bgat  = (const float*)d_in[4];
    const float* wgcn  = (const float*)d_in[5];
    const float* bgcn  = (const float*)d_in[6];
    const float* gamma = (const float*)d_in[7];
    const float* beta  = (const float*)d_in[8];
    const int*   ei    = (const int*)d_in[9];

    const int  n  = in_sizes[0] / DCH;     // 50000
    const int  E  = in_sizes[9] / 2;       // 800000
    const long Et = (long)E + n;           // edges + self loops

    float* ws   = (float*)d_ws;
    // packed f16 weights first (guarantees 32B alignment for v16h loads)
    _Float16* WPK = (_Float16*)ws;         // 3*128*128 halves = 24576 floats
    float* XL   = ws + 24576;              // XL | XR | XW contiguous
    float* XR   = XL  + (size_t)n * DCH;
    float* XW   = XR  + (size_t)n * DCH;
    float* ACC  = XW  + (size_t)n * DCH;
    float* LOG  = ACC + (size_t)n * DCH;
    float* MMAX = LOG + (size_t)Et * NHEAD;
    float* DEN  = MMAX + (size_t)n * NHEAD;
    float* DEG  = DEN  + (size_t)n * NHEAD;
    float* DINV = DEG  + (size_t)n;
    float* SUMS = DINV + (size_t)n;        // 128
    float* SUMQ = SUMS + 128;              // 128
    float* MUSD = SUMQ + 128;              // 256 (mu, rstd)

    const long tot = (long)n * DCH;

    wpack_kernel<<<96, 256, 0, stream>>>(wgl, wgr, wgcn, (unsigned int*)WPK);

    gemm3_wmma_kernel<<<(n + 127) / 128, 256, 0, stream>>>(x, WPK, XL, n);

    init_kernel<<<(int)((tot + 255) / 256), 256, 0, stream>>>(
        ACC, MMAX, DEN, DEG, SUMS, n);

    edge_logits_kernel<<<(int)((Et + 7) / 8), 256, 0, stream>>>(
        ei, E, Et, XL, XR, att, LOG, MMAX, DEG);

    dinv_kernel<<<(n + 255) / 256, 256, 0, stream>>>(DEG, DINV, n);

    edge_denom_kernel<<<(int)((Et * NHEAD + 255) / 256), 256, 0, stream>>>(
        LOG, MMAX, ei, E, Et, DEN);

    edge_scatter_kernel<<<(int)((Et + 7) / 8), 256, 0, stream>>>(
        ei, E, Et, XL, XW, LOG, MMAX, DEN, DINV, ACC);

    bn_pre_kernel<<<(n + 63) / 64, 256, 0, stream>>>(
        ACC, x, bgat, bgcn, SUMS, SUMQ, n);

    bn_stats_kernel<<<1, 128, 0, stream>>>(SUMS, SUMQ, MUSD, 1.0f / (float)n);

    bn_apply_kernel<<<(int)((tot + 255) / 256), 256, 0, stream>>>(
        ACC, MUSD, gamma, beta, (float*)d_out, tot);
}
